// NL_Block_15994458211031
// MI455X (gfx1250) — matrix operand order repaired and checked by reading, not compiled, not run on hardware
//
#include <hip/hip_runtime.h>
#include <hip/hip_bf16.h>
#include <stdint.h>

typedef unsigned short u16;
typedef __attribute__((ext_vector_type(16))) __bf16 v16bf;
typedef __attribute__((ext_vector_type(8)))  float  v8f;

#define BB 4
#define CC 256
#define HH 64
#define WW2 64
#define NN 4096          // H*W
#define KCH 8            // 256 / 32 k-chunks for bf16 WMMA (K=32)

union Frag { v16bf v; uint4 u[2]; };

__device__ inline u16 f2bf(float x) {
    unsigned u = __float_as_uint(x);
    u += 0x7fffu + ((u >> 16) & 1u);       // round-to-nearest-even
    return (u16)(u >> 16);
}

__device__ inline v8f wmma_bf16(v16bf a, v16bf b, v8f c) {
    return __builtin_amdgcn_wmma_f32_16x16x32_bf16(
        false, a, false, b, (short)0, c, false, false);
}

// CDNA5 async global->LDS copy, 16B per lane (tracked by ASYNCcnt)
__device__ inline void async_copy16(uint32_t lds_off, const void* gaddr) {
    asm volatile("global_load_async_to_lds_b128 %0, %1, off"
                 :: "v"(lds_off), "v"(gaddr) : "memory");
}
__device__ inline void wait_async0() { asm volatile("s_wait_asynccnt 0x0" ::: "memory"); }
__device__ inline void wait_async2() { asm volatile("s_wait_asynccnt 0x2" ::: "memory"); }
__device__ inline void wait_async8() { asm volatile("s_wait_asynccnt 0x8" ::: "memory"); }

// ---------------------------------------------------------------------------
// tiny zero pad row (branchless conv halo handling reads from here)
// ---------------------------------------------------------------------------
__global__ void zero_pad_kernel(u16* __restrict__ z) { z[threadIdx.x] = 0; }

// ---------------------------------------------------------------------------
// Prep: weights fp32 [O][I][3][3] -> bf16 B-frag layout [tap][k][co][32ci]
// ---------------------------------------------------------------------------
__global__ void prep_w_kernel(const float* __restrict__ w, u16* __restrict__ wB) {
    unsigned idx = blockIdx.x * 256 + threadIdx.x;
    if (idx >= 9u * 256u * 256u) return;
    int cc  = idx & 31;
    int co  = (idx >> 5) & 255;
    int k   = (idx >> 13) & 7;
    int tap = idx >> 16;                   // 0..8
    int ci  = k * 32 + cc;
    wB[idx] = f2bf(w[((size_t)(co * 256 + ci)) * 9 + tap]);
}

// ---------------------------------------------------------------------------
// Prep: x fp32 [B][C][N] -> xq bf16 [B][N][C] (A-operand, pos-major)
//                        -> xkB bf16 [B][8][N][32] (B-operand frag order)
// ---------------------------------------------------------------------------
__global__ void prep_x_kernel(const float* __restrict__ x,
                              u16* __restrict__ xq, u16* __restrict__ xkB) {
    unsigned idx = blockIdx.x * 256 + threadIdx.x;    // over B*C*N = 2^22
    if (idx >= (unsigned)(BB * CC * NN)) return;
    int n = idx & 4095;
    int c = (idx >> 12) & 255;
    int b = idx >> 20;
    u16 h = f2bf(x[idx]);
    xq[((size_t)(b * NN + n)) * CC + c] = h;
    xkB[(((size_t)(b * 8 + (c >> 5)) * NN + n)) * 32 + (c & 31)] = h;
}

// ---------------------------------------------------------------------------
// conv3x3 as 9 shifted GEMMs.  src bf16 [B][N][C], wB [9][8][256][32],
// dst f32 [B][C][N].  The 16(pos)x256(ci) A slab for each tap is staged into
// LDS once per block with async copies (double-buffered over taps) and shared
// by all 8 waves; halo rows read a zero page (branchless padding).
// Tap loop fully unrolled: kh/kw, buffer ping-pong and wait immediates are
// all compile-time; 72 static WMMAs per wave.
// ---------------------------------------------------------------------------
__device__ inline void stage_tap(const u16* __restrict__ src,
                                 const u16* __restrict__ zrow,
                                 int b, int nb, int tap,
                                 uint32_t ldsbase, int tid) {
    const int kh = tap / 3, kw = tap - kh * 3;
    #pragma unroll
    for (int i = 0; i < 2; ++i) {
        int q = tid + 256 * i;            // chunk 0..511 (16B each) of 8KB slab
        int p = nb + (q >> 5);            // position row (32 chunks per row)
        int h = p >> 6, w = p & 63;
        int h2 = h + kh - 1, w2 = w + kw - 1;
        bool valid = (h2 >= 0) & (h2 < HH) & (w2 >= 0) & (w2 < WW2);
        int n2 = h2 * 64 + w2;
        int cidx = (q & 31) * 8;          // u16 offset within 256-ci row
        const u16* g = valid ? (src + ((size_t)(b * NN + n2)) * CC + cidx)
                             : (zrow + cidx);
        async_copy16(ldsbase + (uint32_t)q * 16, g);
    }
}

__global__ __launch_bounds__(256) void conv3x3_wmma_kernel(
    const u16* __restrict__ src, const u16* __restrict__ wB,
    const u16* __restrict__ zrow, float* __restrict__ dst) {
    __shared__ __align__(16) u16 lds_a[2][4096];   // [pos16][ci256] double buf

    const int b    = blockIdx.z;
    const int tid  = threadIdx.x;
    const int wid  = tid >> 5;
    const int lane = tid & 31;
    const int lh   = lane >> 4;
    const int l16  = lane & 15;
    const int cbase = (blockIdx.y * 8 + wid) * 16;    // co tile base
    const int nb    = blockIdx.x * 16;                // position tile base

    const uint32_t la0 = (uint32_t)(uintptr_t)lds_a[0];
    const uint32_t la1 = (uint32_t)(uintptr_t)lds_a[1];

    v8f acc = {0.f,0.f,0.f,0.f,0.f,0.f,0.f,0.f};

    stage_tap(src, zrow, b, nb, 0, la0, tid);
    #pragma unroll
    for (int tap = 0; tap < 9; ++tap) {
        const u16* cur = lds_a[tap & 1];
        if (tap < 8) {
            stage_tap(src, zrow, b, nb, tap + 1, (tap & 1) ? la0 : la1, tid);
            wait_async2();                // current tap slab ready; next in flight
        } else {
            wait_async0();
        }
        __syncthreads();

        #pragma unroll
        for (int k = 0; k < KCH; ++k) {
            int ci0 = k * 32 + lh * 16;   // A layout: K = lh*16 + {0..15}
            const u16* ar = cur + (l16 * 256 + ci0);
            Frag a; a.u[0] = *(const uint4*)ar; a.u[1] = *(const uint4*)(ar + 8);
            const u16* wp = wB + (((size_t)(tap * 8 + k) * 256 + cbase + l16)) * 32 + lh * 16;
            Frag bf; bf.u[0] = *(const uint4*)wp; bf.u[1] = *(const uint4*)(wp + 8);
            acc = wmma_bf16(a.v, bf.v, acc);
        }
        __syncthreads();                  // before next tap overwrites its buffer
    }

    for (int j = 0; j < 8; ++j) {
        int n  = nb + j + 8 * lh;
        int co = cbase + l16;
        dst[((size_t)(b * CC + co)) * NN + n] = acc[j];
    }
}

// ---------------------------------------------------------------------------
// BN stats: one block per channel, reduce over B*N = 16384
// ---------------------------------------------------------------------------
__global__ __launch_bounds__(256) void bn_stats_kernel(
    const float* __restrict__ pre, float* __restrict__ mean,
    float* __restrict__ invstd) {
    const int c = blockIdx.x;
    __shared__ float s1[256], s2[256];
    float a = 0.f, q = 0.f;
    for (int i = threadIdx.x; i < BB * NN; i += 256) {
        int bb = i >> 12, n = i & 4095;
        float v = pre[((size_t)(bb * CC + c)) * NN + n];
        a += v; q += v * v;
    }
    s1[threadIdx.x] = a; s2[threadIdx.x] = q;
    __syncthreads();
    for (int d = 128; d > 0; d >>= 1) {
        if (threadIdx.x < d) {
            s1[threadIdx.x] += s1[threadIdx.x + d];
            s2[threadIdx.x] += s2[threadIdx.x + d];
        }
        __syncthreads();
    }
    if (threadIdx.x == 0) {
        float m = s1[0] * (1.0f / 16384.0f);
        float var = s2[0] * (1.0f / 16384.0f) - m * m;
        mean[c]   = m;
        invstd[c] = rsqrtf(var + 1e-5f);
    }
}

// BN apply -> bf16 value in [B][C][N] (== B-frag-friendly V layout)
__global__ void bn_value_kernel(const float* __restrict__ pre,
                                const float* __restrict__ mean,
                                const float* __restrict__ invstd,
                                const float* __restrict__ g,
                                const float* __restrict__ beta,
                                u16* __restrict__ vout) {
    unsigned idx = blockIdx.x * 256 + threadIdx.x;
    if (idx >= (unsigned)(BB * CC * NN)) return;
    int c = (idx >> 12) & 255;
    float v = (pre[idx] - mean[c]) * invstd[c] * g[c] + beta[c];
    vout[idx] = f2bf(v);
}

// BN + ReLU -> final fp32 output [B][C][N]
__global__ void bn_relu_kernel(const float* __restrict__ pre,
                               const float* __restrict__ mean,
                               const float* __restrict__ invstd,
                               const float* __restrict__ g,
                               const float* __restrict__ beta,
                               float* __restrict__ out) {
    unsigned idx = blockIdx.x * 256 + threadIdx.x;
    if (idx >= (unsigned)(BB * CC * NN)) return;
    int c = (idx >> 12) & 255;
    float v = (pre[idx] - mean[c]) * invstd[c] * g[c] + beta[c];
    out[idx] = fmaxf(v, 0.0f);
}

// ---------------------------------------------------------------------------
// Flash attention: sim = softmax(Q Kᵀ/16), O = sim·Vᵀ, never materializing sim.
// Block: 8 waves = 4 row-tiles (16 rows) x 2 col-halves (128 of 256 cols).
// K/V slabs double-buffered in LDS via async loads: step i+1 streams while the
// 24 WMMAs of step i run (s_wait_asynccnt 8 waits only for the current slab).
// ---------------------------------------------------------------------------
__device__ inline void stage_kv(const u16* __restrict__ xkB,
                                const u16* __restrict__ vB,
                                int b, int m0,
                                uint32_t ldsk, uint32_t ldsv, int tid) {
    #pragma unroll
    for (int i = 0; i < 4; ++i) {
        int li = tid + 256 * i;           // 0..1023, 8 halves (16B) each
        int e0 = li * 8;
        int cc = e0 & 31, mm = (e0 >> 5) & 31, k = e0 >> 10;
        async_copy16(ldsk + (uint32_t)li * 16,
            xkB + (((size_t)(b * 8 + k) * NN + m0 + mm)) * 32 + cc);
        int mv = e0 & 31, c = e0 >> 5;
        async_copy16(ldsv + (uint32_t)li * 16,
            vB + ((size_t)(b * CC + c)) * NN + m0 + mv);
    }
}

__global__ __launch_bounds__(256) void flash_attn_kernel(
    const u16* __restrict__ xq,   // [B][N][C]
    const u16* __restrict__ xkB,  // [B][8][N][32]
    const u16* __restrict__ vB,   // [B][C][N]
    u16* __restrict__ ctx) {      // [B][N][C]
    __shared__ __align__(16) u16 lds_k[2][8192];  // [k8][m32][cc32] double buf
    __shared__ __align__(16) u16 lds_v[2][8192];  // [c256][m32]     double buf
    __shared__ __align__(16) u16 lds_p[4][512];   // per row-tile: [row16][m32]

    const int b    = blockIdx.y;
    const int tid  = threadIdx.x;
    const int wid  = tid >> 5;
    const int lane = tid & 31;
    const int lh   = lane >> 4;
    const int l16  = lane & 15;
    const int rt   = wid >> 1;            // row tile in block
    const int ch   = wid & 1;             // col half
    const int cbase = ch * 128;
    const int nb    = (blockIdx.x * 4 + rt) * 16;

    const uint32_t lk[2] = { (uint32_t)(uintptr_t)lds_k[0], (uint32_t)(uintptr_t)lds_k[1] };
    const uint32_t lv[2] = { (uint32_t)(uintptr_t)lds_v[0], (uint32_t)(uintptr_t)lds_v[1] };

    // Q fragments (16 rows x 256 cols, 8 k-chunks) kept resident
    v16bf qf[KCH];
    {
        const u16* qrow = xq + ((size_t)(b * NN + nb + l16)) * CC;
        #pragma unroll
        for (int k = 0; k < KCH; ++k) {
            int ci0 = k * 32 + lh * 16;
            Frag f;
            f.u[0] = *(const uint4*)(qrow + ci0);
            f.u[1] = *(const uint4*)(qrow + ci0 + 8);
            qf[k] = f.v;
        }
    }

    const v8f vzero = {0.f,0.f,0.f,0.f,0.f,0.f,0.f,0.f};
    v8f acc[8];
    float mrow[8], lrow[8];
    for (int t = 0; t < 8; ++t) acc[t] = vzero;
    for (int j = 0; j < 8; ++j) { mrow[j] = -1e30f; lrow[j] = 0.0f; }

    const float scale = 0.0625f;          // C^-0.5 = 1/16

    stage_kv(xkB, vB, b, 0, lk[0], lv[0], tid);
    for (int si = 0; si < NN / 32; ++si) {
        const int m0   = si * 32;
        const int cur  = si & 1;
        const int nxtm = (m0 + 32 < NN) ? (m0 + 32) : m0;   // tail: benign reload
        stage_kv(xkB, vB, b, nxtm, lk[cur ^ 1], lv[cur ^ 1], tid);
        wait_async8();                    // current slab landed; next in flight
        __syncthreads();

        const u16* kbuf = lds_k[cur];
        const u16* vbuf = lds_v[cur];

        // ---- S = Q · K tile (two 16-key tiles) ----
        v8f S0 = vzero, S1 = vzero;
        #pragma unroll
        for (int k = 0; k < KCH; ++k) {
            const u16* p0 = kbuf + ((k * 32 + l16) * 32 + lh * 16);
            Frag f0; f0.u[0] = *(const uint4*)p0; f0.u[1] = *(const uint4*)(p0 + 8);
            S0 = wmma_bf16(qf[k], f0.v, S0);
            const u16* p1 = kbuf + ((k * 32 + 16 + l16) * 32 + lh * 16);
            Frag f1; f1.u[0] = *(const uint4*)p1; f1.u[1] = *(const uint4*)(p1 + 8);
            S1 = wmma_bf16(qf[k], f1.v, S1);
        }

        // ---- online softmax (row reductions across 16-lane groups) ----
        float p0v[8], p1v[8], alpha[8];
        #pragma unroll
        for (int j = 0; j < 8; ++j) {
            float s0 = S0[j] * scale, s1 = S1[j] * scale;
            float mx = fmaxf(s0, s1);
            for (int d = 1; d < 16; d <<= 1) mx = fmaxf(mx, __shfl_xor(mx, d, 32));
            float mn = fmaxf(mrow[j], mx);
            alpha[j] = __expf(mrow[j] - mn);
            mrow[j]  = mn;
            p0v[j] = __expf(s0 - mn);
            p1v[j] = __expf(s1 - mn);
            float rs = p0v[j] + p1v[j];
            for (int d = 1; d < 16; d <<= 1) rs += __shfl_xor(rs, d, 32);
            lrow[j] = lrow[j] * alpha[j] + rs;
        }
        #pragma unroll
        for (int t = 0; t < 8; ++t)
            #pragma unroll
            for (int j = 0; j < 8; ++j) acc[t][j] *= alpha[j];

        // ---- P: C-layout -> A-layout via LDS transpose (half 0 writes) ----
        if (ch == 0) {
            #pragma unroll
            for (int j = 0; j < 8; ++j) {
                int row = j + 8 * lh;
                lds_p[rt][row * 32 + l16]      = f2bf(p0v[j]);
                lds_p[rt][row * 32 + 16 + l16] = f2bf(p1v[j]);
            }
        }
        __syncthreads();

        Frag pf;
        {
            const u16* pr = lds_p[rt] + (l16 * 32 + lh * 16);
            pf.u[0] = *(const uint4*)pr;
            pf.u[1] = *(const uint4*)(pr + 8);
        }
        // ---- O += P · Vᵀ tile (8 col tiles of 16) ----
        #pragma unroll
        for (int t = 0; t < 8; ++t) {
            int c = cbase + t * 16 + l16;
            const u16* pv = vbuf + (c * 32 + lh * 16);
            Frag vf; vf.u[0] = *(const uint4*)pv; vf.u[1] = *(const uint4*)(pv + 8);
            acc[t] = wmma_bf16(pf.v, vf.v, acc[t]);
        }
        __syncthreads();                  // cur buffers free for overwrite
    }

    // epilogue: O/l -> ctx bf16 [B][N][C] (pos-major: ready for conv2 A-loads)
    for (int j = 0; j < 8; ++j) {
        int n = nb + j + 8 * lh;
        float inv = 1.0f / lrow[j];
        u16* orow = ctx + ((size_t)(b * NN + n)) * CC;
        for (int t = 0; t < 8; ++t)
            orow[cbase + t * 16 + l16] = f2bf(acc[t][j] * inv);
    }
}

// ---------------------------------------------------------------------------
extern "C" void kernel_launch(void* const* d_in, const int* in_sizes, int n_in,
                              void* d_out, int out_size, void* d_ws, size_t ws_size,
                              hipStream_t stream) {
    const float* x     = (const float*)d_in[0];
    const float* wv    = (const float*)d_in[1];
    // bv (d_in[2]) is exactly cancelled by training-mode BN -> unused
    const float* gv    = (const float*)d_in[3];
    const float* betav = (const float*)d_in[4];
    const float* ww    = (const float*)d_in[5];
    // bw (d_in[6]) cancelled likewise
    const float* gw    = (const float*)d_in[7];
    const float* betaw = (const float*)d_in[8];
    float* out = (float*)d_out;

    char*  ws  = (char*)d_ws;
    size_t off = 0;
    auto alloc = [&](size_t bytes) -> void* {
        void* p = ws + off;
        off += (bytes + 255) & ~size_t(255);
        return p;
    };
    const size_t ELEMS = (size_t)BB * CC * NN;
    u16*   xq     = (u16*)  alloc(ELEMS * 2);                 // [B][N][C] bf16
    u16*   xkB    = (u16*)  alloc(ELEMS * 2);                 // [B][8][N][32] bf16
    u16*   wvB    = (u16*)  alloc((size_t)9 * 256 * 256 * 2);
    u16*   wwB    = (u16*)  alloc((size_t)9 * 256 * 256 * 2);
    float* v_pre  = (float*)alloc(ELEMS * 4);                 // conv1 out [B][C][N]
    float* mean_v = (float*)alloc(256 * 4);
    float* istd_v = (float*)alloc(256 * 4);
    u16*   valB   = (u16*)  alloc(ELEMS * 2);                 // value bf16 [B][C][N]
    u16*   ctx    = (u16*)  alloc(ELEMS * 2);                 // attn out [B][N][C]
    float* w_pre  = (float*)alloc(ELEMS * 4);                 // conv2 out [B][C][N]
    float* mean_w = (float*)alloc(256 * 4);
    float* istd_w = (float*)alloc(256 * 4);
    u16*   zrow   = (u16*)  alloc(256 * 2);                   // zero pad row

    // 0) pack operands to bf16 WMMA layouts; zero the pad row (ws is poisoned)
    zero_pad_kernel<<<1, 256, 0, stream>>>(zrow);
    prep_w_kernel<<<(9 * 256 * 256 + 255) / 256, 256, 0, stream>>>(wv, wvB);
    prep_w_kernel<<<(9 * 256 * 256 + 255) / 256, 256, 0, stream>>>(ww, wwB);
    prep_x_kernel<<<(unsigned)(ELEMS / 256), 256, 0, stream>>>(x, xq, xkB);

    // 1) value branch conv (bias cancels under BN)
    conv3x3_wmma_kernel<<<dim3(256, 2, BB), 256, 0, stream>>>(xq, wvB, zrow, v_pre);
    bn_stats_kernel<<<256, 256, 0, stream>>>(v_pre, mean_v, istd_v);
    bn_value_kernel<<<(unsigned)(ELEMS / 256), 256, 0, stream>>>(
        v_pre, mean_v, istd_v, gv, betav, valB);

    // 2) fused attention (softmax(XᵀX/√C) · valueᵀ), flash style
    flash_attn_kernel<<<dim3(64, BB), 256, 0, stream>>>(xq, xkB, valB, ctx);

    // 3) output conv + BN + ReLU
    conv3x3_wmma_kernel<<<dim3(256, 2, BB), 256, 0, stream>>>(ctx, wwB, zrow, w_pre);
    bn_stats_kernel<<<256, 256, 0, stream>>>(w_pre, mean_w, istd_w);
    bn_relu_kernel<<<(unsigned)(ELEMS / 256), 256, 0, stream>>>(
        w_pre, mean_w, istd_w, gw, betaw, out);
}
